// SimpleSNN_81612968558815
// MI455X (gfx1250) — matrix operand order repaired and checked by reading, not compiled, not run on hardware
//
#include <hip/hip_runtime.h>

typedef __attribute__((ext_vector_type(16))) _Float16 v16h;
typedef __attribute__((ext_vector_type(8)))  float    v8f;

constexpr float BETA = 0.9f;
constexpr float THR  = 1.0f;
constexpr int   NSTEPS = 300;
constexpr int   BATCH  = 8192;
constexpr int   FEAT   = 784;   // = 24*32 + 16
constexpr int   HID    = 128;
constexpr int   OUT    = 10;
constexpr int   LDS_STRIDE = 132; // 128 + 4 pad -> conflict-free column reads

__global__ __launch_bounds__(128)
void snn_fused(const float* __restrict__ x,
               const float* __restrict__ W1,
               const float* __restrict__ b1,
               const float* __restrict__ W2,
               const float* __restrict__ b2,
               float* __restrict__ spk_out,   // [300][8192][10]
               float* __restrict__ mem_out)   // [300][8192][10]
{
    __shared__ float lds[4][16][LDS_STRIDE];

    const int lane  = threadIdx.x & 31;
    const int wid   = threadIdx.x >> 5;
    const int tile  = blockIdx.x * 4 + wid;   // 0..511
    const int mbase = tile * 16;
    const int half  = lane >> 4;              // 0: lanes 0-15, 1: lanes 16-31
    const int l16   = lane & 15;

    // ---------------- GEMM1: cur1 = x @ W1^T + b1 (f16 WMMA, f32 acc) -----
    v8f acc[8];
    #pragma unroll
    for (int nt = 0; nt < 8; ++nt) acc[nt] = v8f{};

    const float* xrow = x + (size_t)(mbase + l16) * FEAT;

    // 24 full K-chunks of 32
    for (int kc = 0; kc < 24; ++kc) {
        const int ka = kc * 32 + half * 8;       // A: interleaved 8+8 layout
        v16h a;
        #pragma unroll
        for (int i = 0; i < 8; ++i) {
            a[i]     = (_Float16)xrow[ka + i];
            a[8 + i] = (_Float16)xrow[ka + 16 + i];
        }
        const int kb = kc * 32 + half * 8;       // B: interleaved 8+8 layout
        #pragma unroll
        for (int nt = 0; nt < 8; ++nt) {
            const float* wrow = W1 + (size_t)(nt * 16 + l16) * FEAT;
            v16h bf;
            #pragma unroll
            for (int i = 0; i < 16; ++i) bf[i] = (_Float16)wrow[kb + (i < 8 ? i : 8 + i)];
            acc[nt] = __builtin_amdgcn_wmma_f32_16x16x32_f16(
                false, a, false, bf, (short)0, acc[nt], false, false);
        }
    }
    // Tail chunk: K = 768..783 valid, rest zero
    {
        const int ka = 768 + half * 8;           // ka+7 <= 783: valid
        v16h a;
        #pragma unroll
        for (int i = 0; i < 8; ++i) {
            a[i]     = (_Float16)xrow[ka + i];
            a[8 + i] = (_Float16)0.0f;           // K >= 784
        }
        #pragma unroll
        for (int nt = 0; nt < 8; ++nt) {
            const float* wrow = W1 + (size_t)(nt * 16 + l16) * FEAT;
            v16h bf;
            #pragma unroll
            for (int i = 0; i < 16; ++i) {
                float v = wrow[768 + half * 8 + (i & 7)]; // always in-bounds read
                bf[i] = (i >= 8) ? (_Float16)0.0f : (_Float16)v; // i>=8 -> K>=784
            }
            acc[nt] = __builtin_amdgcn_wmma_f32_16x16x32_f16(
                false, a, false, bf, (short)0, acc[nt], false, false);
        }
    }

    // Add b1, stage C-layout tiles to LDS for layout conversion
    #pragma unroll
    for (int nt = 0; nt < 8; ++nt) {
        const float b1v = b1[nt * 16 + l16];
        #pragma unroll
        for (int r = 0; r < 8; ++r)
            lds[wid][r + half * 8][nt * 16 + l16] = acc[nt][r] + b1v;
    }
    __syncthreads();

    // Read cur1 back in WMMA A-fragment layout (per K-chunk of 32)
    float cur1[64];
    #pragma unroll
    for (int c = 0; c < 4; ++c) {
        const int j0 = c * 32 + half * 8;
        #pragma unroll
        for (int i = 0; i < 8; ++i) {
            cur1[c * 16 + i]     = lds[wid][l16][j0 + i];
            cur1[c * 16 + 8 + i] = lds[wid][l16][j0 + 16 + i];
        }
    }

    // Resident W2 B-fragments (B[k][n] = W2[n][k]; n = l16, zero-pad n>=10)
    v16h bw2[4];
    #pragma unroll
    for (int c = 0; c < 4; ++c) {
        const int j0 = c * 32 + half * 8;
        #pragma unroll
        for (int i = 0; i < 16; ++i) {
            float v = (l16 < OUT) ? W2[l16 * HID + j0 + (i < 8 ? i : 8 + i)] : 0.0f;
            bw2[c][i] = (_Float16)v;
        }
    }
    // b2 broadcast held as the persistent WMMA C-initializer
    v8f b2vec;
    {
        const float b2v = (l16 < OUT) ? b2[l16] : 0.0f;
        #pragma unroll
        for (int r = 0; r < 8; ++r) b2vec[r] = b2v;
    }

    // Persistent state: membranes + previous-step spikes (exact 0/1)
    float mem1[64];
    #pragma unroll
    for (int i = 0; i < 64; ++i) mem1[i] = 0.0f;
    v16h spk[4];                    // layer-1 spikes, WMMA A-fragment layout
    #pragma unroll
    for (int c = 0; c < 4; ++c)
        #pragma unroll
        for (int i = 0; i < 16; ++i) spk[c][i] = (_Float16)0.0f;
    v8f mem2  = v8f{};
    v8f spk2f = v8f{};              // layer-2 spikes (f32, doubles as store value)

    float* sp = spk_out + (size_t)mbase * OUT;
    float* mp = mem_out + (size_t)mbase * OUT;

    for (int t = 0; t < NSTEPS; ++t) {
        v8f c2 = b2vec;             // C-init carries the bias

        #pragma unroll
        for (int c = 0; c < 4; ++c) {
            v16h ns;
            #pragma unroll
            for (int i = 0; i < 16; ++i) {
                const int idx = c * 16 + i;
                // reset(t) == spk(t-1), reuse stored spike: 1 cvt vs cmp+sel
                const float sel = (float)spk[c][i];
                float m = __builtin_fmaf(BETA, mem1[idx], cur1[idx]) - sel;
                mem1[idx] = m;
                ns[i] = (m > THR) ? (_Float16)1.0f : (_Float16)0.0f;
            }
            spk[c] = ns;
            c2 = __builtin_amdgcn_wmma_f32_16x16x32_f16(
                false, spk[c], false, bw2[c], (short)0, c2, false, false);
        }

        // Layer-2 LIF + record (lanes with n >= 10 carry dead values)
        if (l16 < OUT) {
            #pragma unroll
            for (int r = 0; r < 8; ++r) {
                float m = __builtin_fmaf(BETA, mem2[r], c2[r]) - spk2f[r];
                mem2[r] = m;
                const float s = (m > THR) ? 1.0f : 0.0f;
                spk2f[r] = s;
                const int off = (r + half * 8) * OUT + l16;
                sp[off] = s;
                mp[off] = m;
            }
        }
        sp += (size_t)BATCH * OUT;
        mp += (size_t)BATCH * OUT;
    }
}

extern "C" void kernel_launch(void* const* d_in, const int* in_sizes, int n_in,
                              void* d_out, int out_size, void* d_ws, size_t ws_size,
                              hipStream_t stream) {
    const float* x  = (const float*)d_in[0];
    const float* W1 = (const float*)d_in[1];
    const float* b1 = (const float*)d_in[2];
    const float* W2 = (const float*)d_in[3];
    const float* b2 = (const float*)d_in[4];
    float* out = (float*)d_out;
    float* spk = out;
    float* mem = out + (size_t)NSTEPS * BATCH * OUT;

    // 512 batch tiles of 16 rows; 4 waves per block -> 128 blocks x 128 threads
    snn_fused<<<128, 128, 0, stream>>>(x, W1, b1, W2, b2, spk, mem);
}
